// TransformerBlock_26860725469658
// MI455X (gfx1250) — matrix operand order repaired and checked by reading, not compiled, not run on hardware
//
#include <hip/hip_runtime.h>
#include <cstdint>

// ---------------------------------------------------------------------------
// Types / WMMA wrapper (CDNA5 gfx1250, wave32)
// ---------------------------------------------------------------------------
typedef __bf16 bf16;
typedef __attribute__((ext_vector_type(16))) __bf16 bf16x16;
typedef __attribute__((ext_vector_type(8)))  float  floatx8;

union Frag { bf16x16 v; uint4 u[2]; };

__device__ __forceinline__ floatx8 wmma_bf16(bf16x16 a, bf16x16 b, floatx8 c) {
  // 8 args: (neg_a, A, neg_b, B, c_mod, C, reuse_a, reuse_b)
  return __builtin_amdgcn_wmma_f32_16x16x32_bf16(false, a, false, b, (short)0, c,
                                                 false, false);
}

constexpr int TOK = 8192;    // B*S
constexpr int HD  = 1024;    // H
constexpr int TH3 = 3072;    // 3H
constexpr int FD  = 4096;    // F_DIM
constexpr int NE  = 8;       // experts
constexpr int NHh = 16;      // heads
constexpr int DHh = 64;      // head dim
constexpr int Sseq = 1024;
constexpr int CAP = 17408;   // 16384 assignments + 8*128 padding

// ---------------------------------------------------------------------------
// fp32 -> bf16 converts (plain + transposed, for W in [N,K] row-major layout)
// ---------------------------------------------------------------------------
__global__ void f2b_kernel(const float* __restrict__ in, bf16* __restrict__ out, long n) {
  long i = (long)blockIdx.x * blockDim.x + threadIdx.x;
  long st = (long)gridDim.x * blockDim.x;
  for (; i < n; i += st) out[i] = (bf16)in[i];
}

// in [E][R][C] -> out [E][C][R]
__global__ void f2b_tr_kernel(const float* __restrict__ in, bf16* __restrict__ out,
                              int E, int R, int C) {
  long n = (long)E * R * C;
  long i = (long)blockIdx.x * blockDim.x + threadIdx.x;
  long st = (long)gridDim.x * blockDim.x;
  for (; i < n; i += st) {
    long e = i / ((long)R * C);
    long rem = i - e * (long)R * C;
    long r = rem / C;
    long c = rem - r * C;
    out[(e * (long)C + c) * R + r] = (bf16)in[i];
  }
}

// ---------------------------------------------------------------------------
// LayerNorm (fp32 in, bf16 out), one block per token row
// ---------------------------------------------------------------------------
__global__ __launch_bounds__(256) void ln_kernel(const float* __restrict__ x,
                                                 const float* __restrict__ w,
                                                 const float* __restrict__ b,
                                                 bf16* __restrict__ out) {
  __shared__ float r1[256], r2[256];
  long row = blockIdx.x;
  const float* xr = x + row * HD;
  int tid = threadIdx.x;
  float s = 0.f, s2 = 0.f;
  for (int i = tid; i < HD; i += 256) { float v = xr[i]; s += v; s2 += v * v; }
  r1[tid] = s; r2[tid] = s2;
  __syncthreads();
  for (int st = 128; st > 0; st >>= 1) {
    if (tid < st) { r1[tid] += r1[tid + st]; r2[tid] += r2[tid + st]; }
    __syncthreads();
  }
  float mu  = r1[0] * (1.0f / HD);
  float var = r2[0] * (1.0f / HD) - mu * mu;
  float inv = rsqrtf(var + 1e-5f);
  for (int i = tid; i < HD; i += 256)
    out[row * HD + i] = (bf16)(((xr[i] - mu) * inv) * w[i] + b[i]);
}

// ---------------------------------------------------------------------------
// Tiled WMMA GEMM:  C[M,N] = A[M,K] @ Bm[N,K]^T  (+ epilogue)
// Block tile 128x128, K-step 32, 8 waves as 4(M)x2(N), wave tile 32x64.
// EPI: 0 = +bias -> bf16 (QKV)
//      1 = +bias +resid -> fp32 (out-proj residual)
//      2 = +bias, GELU -> bf16 (MoE hidden)
//      3 = (+bias) * rowscale -> fp32 (MoE expert out, prob-scaled)
// EXPERT: blockIdx.z selects expert; rows come from padded offsets exp_off[].
// ---------------------------------------------------------------------------
constexpr int LDT = 48;  // LDS row stride in halves (96B, multiple of 16B)

template <int EPI, bool EXPERT>
__global__ __launch_bounds__(256) void wmma_gemm(
    const bf16* __restrict__ A, const bf16* __restrict__ Bm,
    const float* __restrict__ bias, const float* __restrict__ resid,
    const float* __restrict__ rowscale,
    float* __restrict__ Cf, bf16* __restrict__ Ch,
    int N, int K, const int* __restrict__ exp_off, long wstride, long bstride) {
  __shared__ __align__(16) bf16 As[128 * LDT];
  __shared__ __align__(16) bf16 Bs[128 * LDT];

  int tid = threadIdx.x;
  int n0 = blockIdx.x * 128;
  int m0 = blockIdx.y * 128;
  long arow0 = 0;
  if constexpr (EXPERT) {
    int e = blockIdx.z;
    int r0 = exp_off[e], r1 = exp_off[e + 1];
    if (m0 >= (r1 - r0)) return;  // block-uniform exit: EXEC stays all-ones
    arow0 = r0;
    A    += (long)r0 * K;
    Bm   += (long)e * wstride;
    bias += (long)e * bstride;
  }
  int w = tid >> 5, lane = tid & 31;
  int wm = (w & 3) * 32;   // wave M offset in block tile
  int wn = (w >> 2) * 64;  // wave N offset
  int mrow = lane & 15;
  int kb = (lane < 16) ? 0 : 8;   // A-fragment K base
  int ks = (lane < 16) ? 0 : 8;   // B-fragment K base

  floatx8 acc[2][4];
#pragma unroll
  for (int i = 0; i < 2; ++i)
#pragma unroll
    for (int j = 0; j < 4; ++j) acc[i][j] = (floatx8){0, 0, 0, 0, 0, 0, 0, 0};

  for (int k0 = 0; k0 < K; k0 += 32) {
#pragma unroll
    for (int i = 0; i < 2; ++i) {  // 512 uint4 per tile / 256 threads
      int u = tid + i * 256;
      int row = u >> 2;
      int col = (u & 3) * 8;
      *(uint4*)&As[row * LDT + col] =
          *(const uint4*)(A + (long)(m0 + row) * K + k0 + col);
      *(uint4*)&Bs[row * LDT + col] =
          *(const uint4*)(Bm + (long)(n0 + row) * K + k0 + col);
    }
    __syncthreads();
    bf16x16 fa[2];
#pragma unroll
    for (int i = 0; i < 2; ++i) {
      Frag f;
      const bf16* p = &As[(wm + i * 16 + mrow) * LDT + kb];
      f.u[0] = *(const uint4*)p;
      f.u[1] = *(const uint4*)(p + 16);
      fa[i] = f.v;
    }
#pragma unroll
    for (int j = 0; j < 4; ++j) {
      Frag fb;
      const bf16* p = &Bs[(wn + j * 16 + mrow) * LDT + ks];
      fb.u[0] = *(const uint4*)p;
      fb.u[1] = *(const uint4*)(p + 16);
#pragma unroll
      for (int i = 0; i < 2; ++i) acc[i][j] = wmma_bf16(fa[i], fb.v, acc[i][j]);
    }
    __syncthreads();
  }

  int rb = (lane >= 16) ? 8 : 0;
#pragma unroll
  for (int i = 0; i < 2; ++i)
#pragma unroll
    for (int j = 0; j < 4; ++j) {
      long colg = n0 + wn + j * 16 + mrow;
      float bv = bias[colg];
#pragma unroll
      for (int r = 0; r < 8; ++r) {
        long grow = arow0 + m0 + wm + i * 16 + rb + r;
        float v = acc[i][j][r] + bv;
        if constexpr (EPI == 0) {
          Ch[grow * N + colg] = (bf16)v;
        } else if constexpr (EPI == 1) {
          Cf[grow * N + colg] = resid[grow * N + colg] + v;
        } else if constexpr (EPI == 2) {
          v = 0.5f * v * (1.0f + erff(v * 0.70710678118f));
          Ch[grow * N + colg] = (bf16)v;
        } else {
          Cf[grow * N + colg] = v * rowscale[grow];
        }
      }
    }
}

// ---------------------------------------------------------------------------
// Flash attention: block = (qblock of 128 rows, head, batch); 8 waves,
// wave w owns Q rows [w*16, w*16+16). K/V tiles of 128 keys in LDS
// (V transposed for contiguous B-fragments), online softmax via 16-lane
// shfl_xor reductions (stats stay inside each wave32 half).
// ---------------------------------------------------------------------------
__global__ __launch_bounds__(256) void attn_kernel(const bf16* __restrict__ qkv,
                                                   bf16* __restrict__ ctx) {
  __shared__ __align__(16) bf16 Kt[128 * 80];        // [key][dh], stride 80
  __shared__ __align__(16) bf16 Vt[64 * 136];        // [dh][key], stride 136
  __shared__ __align__(16) bf16 Pt[8 * 16 * 136];    // per-wave P tile [16][128]

  int qb = blockIdx.x, h = blockIdx.y, b = blockIdx.z;
  int tid = threadIdx.x, w = tid >> 5, lane = tid & 31;
  int mrow = lane & 15;
  int kb = (lane < 16) ? 0 : 8;
  int ks = (lane < 16) ? 0 : 8;
  int rb = (lane >= 16) ? 8 : 0;
  int q0 = qb * 128;

  const bf16* qptr = qkv + (long)b * Sseq * TH3 + h * DHh;
  const bf16* kptr = qptr + HD;
  const bf16* vptr = qptr + 2 * HD;

  // Q fragments for this wave (rows q0+w*16+mrow, dh split in 2 K-steps)
  bf16x16 qa[2];
  {
    const bf16* base = qptr + (long)(q0 + w * 16 + mrow) * TH3;
#pragma unroll
    for (int t = 0; t < 2; ++t) {
      Frag f;
      const bf16* p = base + t * 32 + kb;
      f.u[0] = *(const uint4*)p;
      f.u[1] = *(const uint4*)(p + 16);
      qa[t] = f.v;
    }
  }

  float mrun[8], lrun[8];
  floatx8 o[4];
#pragma unroll
  for (int r = 0; r < 8; ++r) { mrun[r] = -1e30f; lrun[r] = 0.f; }
#pragma unroll
  for (int n = 0; n < 4; ++n) o[n] = (floatx8){0, 0, 0, 0, 0, 0, 0, 0};

  for (int kt = 0; kt < Sseq / 128; ++kt) {
    int kbase = kt * 128;
    __syncthreads();  // previous iteration's LDS reads done
#pragma unroll
    for (int i = 0; i < 4; ++i) {  // 1024 uint4 per tile / 256 threads
      int u = tid * 4 + i;
      int row = u >> 3;
      int col = (u & 7) * 8;
      *(uint4*)&Kt[row * 80 + col] =
          *(const uint4*)(kptr + (long)(kbase + row) * TH3 + col);
      union { uint4 u4; bf16 hv[8]; } fv;
      fv.u4 = *(const uint4*)(vptr + (long)(kbase + row) * TH3 + col);
#pragma unroll
      for (int e2 = 0; e2 < 8; ++e2) Vt[(col + e2) * 136 + row] = fv.hv[e2];
    }
    __syncthreads();

    // scores S = Q K^T * 1/sqrt(DH)
    floatx8 sc[8];
#pragma unroll
    for (int j = 0; j < 8; ++j) {
      floatx8 a = (floatx8){0, 0, 0, 0, 0, 0, 0, 0};
#pragma unroll
      for (int t = 0; t < 2; ++t) {
        Frag fb;
        const bf16* p = &Kt[(j * 16 + mrow) * 80 + t * 32 + ks];
        fb.u[0] = *(const uint4*)p;
        fb.u[1] = *(const uint4*)(p + 16);
        a = wmma_bf16(qa[t], fb.v, a);
      }
      sc[j] = a;
    }
#pragma unroll
    for (int j = 0; j < 8; ++j)
#pragma unroll
      for (int r = 0; r < 8; ++r) sc[j][r] *= 0.125f;

    // online softmax
#pragma unroll
    for (int r = 0; r < 8; ++r) {
      float mx = -1e30f;
#pragma unroll
      for (int j = 0; j < 8; ++j) mx = fmaxf(mx, sc[j][r]);
      mx = fmaxf(mx, __shfl_xor(mx, 1, 32));
      mx = fmaxf(mx, __shfl_xor(mx, 2, 32));
      mx = fmaxf(mx, __shfl_xor(mx, 4, 32));
      mx = fmaxf(mx, __shfl_xor(mx, 8, 32));
      float mnew = fmaxf(mrun[r], mx);
      float cf = expf(mrun[r] - mnew);
      lrun[r] *= cf;
#pragma unroll
      for (int n = 0; n < 4; ++n) o[n][r] *= cf;
      mrun[r] = mnew;
      float ls = 0.f;
#pragma unroll
      for (int j = 0; j < 8; ++j) {
        float pv = expf(sc[j][r] - mnew);
        Pt[(w * 16 + rb + r) * 136 + j * 16 + mrow] = (bf16)pv;
        ls += pv;
      }
      ls += __shfl_xor(ls, 1, 32);
      ls += __shfl_xor(ls, 2, 32);
      ls += __shfl_xor(ls, 4, 32);
      ls += __shfl_xor(ls, 8, 32);
      lrun[r] += ls;
    }

    // O += P @ V   (wave-private Pt; DS ops are in-order within a wave)
#pragma unroll
    for (int t = 0; t < 4; ++t) {
      Frag fp;
      const bf16* p = &Pt[(w * 16 + mrow) * 136 + t * 32 + kb];
      fp.u[0] = *(const uint4*)p;
      fp.u[1] = *(const uint4*)(p + 16);
#pragma unroll
      for (int n = 0; n < 4; ++n) {
        Frag fv2;
        const bf16* pv = &Vt[(n * 16 + mrow) * 136 + t * 32 + ks];
        fv2.u[0] = *(const uint4*)pv;
        fv2.u[1] = *(const uint4*)(pv + 16);
        o[n] = wmma_bf16(fp.v, fv2.v, o[n]);
      }
    }
  }

  bf16* cbase = ctx + ((long)(b * Sseq + q0 + w * 16)) * HD + h * DHh;
#pragma unroll
  for (int n = 0; n < 4; ++n)
#pragma unroll
    for (int r = 0; r < 8; ++r)
      cbase[(long)(rb + r) * HD + n * 16 + mrow] = (bf16)(o[n][r] / lrun[r]);
}

// ---------------------------------------------------------------------------
// MoE gate: logits, top-2, softmax probs, expert occupancy counting
// ---------------------------------------------------------------------------
__global__ __launch_bounds__(256) void gate_kernel(
    const bf16* __restrict__ xn2, const float* __restrict__ gw,
    const float* __restrict__ gb, int* __restrict__ tk_idx,
    float* __restrict__ tk_p, int* __restrict__ count) {
  __shared__ float red[256 * NE];
  int t = blockIdx.x, tid = threadIdx.x;
  float acc[NE];
#pragma unroll
  for (int e = 0; e < NE; ++e) acc[e] = 0.f;
  const bf16* xr = xn2 + (long)t * HD;
  for (int i = tid; i < HD; i += 256) {
    float xv = (float)xr[i];
#pragma unroll
    for (int e = 0; e < NE; ++e) acc[e] += xv * gw[e * HD + i];
  }
#pragma unroll
  for (int e = 0; e < NE; ++e) red[tid * NE + e] = acc[e];
  __syncthreads();
  for (int s = 128; s > 0; s >>= 1) {
    if (tid < s)
      for (int e = 0; e < NE; ++e) red[tid * NE + e] += red[(tid + s) * NE + e];
    __syncthreads();
  }
  if (tid == 0) {
    float l[NE];
    for (int e = 0; e < NE; ++e) l[e] = red[e] + gb[e];
    int i0 = 0;
    for (int e = 1; e < NE; ++e)
      if (l[e] > l[i0]) i0 = e;
    int i1 = (i0 == 0) ? 1 : 0;
    for (int e = 0; e < NE; ++e)
      if (e != i0 && l[e] > l[i1]) i1 = e;
    float d = expf(l[i1] - l[i0]);
    float inv = 1.0f / (1.0f + d);
    tk_idx[t * 2] = i0;  tk_idx[t * 2 + 1] = i1;
    tk_p[t * 2] = inv;   tk_p[t * 2 + 1] = d * inv;
    atomicAdd(&count[i0], 1);
    atomicAdd(&count[i1], 1);
  }
}

// padded per-expert offsets + cursors (tiny, single thread)
__global__ void scan_kernel(const int* __restrict__ count, int* __restrict__ off,
                            int* __restrict__ cursor) {
  if (blockIdx.x == 0 && threadIdx.x == 0) {
    int o = 0;
    for (int e = 0; e < NE; ++e) {
      off[e] = o;
      cursor[e] = o;
      o += ((count[e] + 127) / 128) * 128;
    }
    off[NE] = o;
  }
}

__global__ __launch_bounds__(256) void place_kernel(
    const int* __restrict__ tk_idx, const float* __restrict__ tk_p,
    int* __restrict__ cursor, int* __restrict__ rows_of,
    int* __restrict__ row_tok, float* __restrict__ row_p) {
  int t = blockIdx.x * 256 + threadIdx.x;
  if (t >= TOK) return;
  for (int k = 0; k < 2; ++k) {
    int e = tk_idx[t * 2 + k];
    int pos = atomicAdd(&cursor[e], 1);
    rows_of[t * 2 + k] = pos;
    row_tok[pos] = t;
    row_p[pos] = tk_p[t * 2 + k];
  }
}

__global__ __launch_bounds__(128) void gather_kernel(const bf16* __restrict__ xn2,
                                                     const int* __restrict__ row_tok,
                                                     bf16* __restrict__ Xg) {
  long r = blockIdx.x;
  int t = row_tok[r];
  const uint4* src = (const uint4*)(xn2 + (long)t * HD);
  uint4* dst = (uint4*)(Xg + r * HD);
  dst[threadIdx.x] = src[threadIdx.x];  // 128 threads * 8 bf16 = 1024
}

__global__ __launch_bounds__(256) void combine_kernel(
    const float* __restrict__ x1, const float* __restrict__ eo,
    const int* __restrict__ rows_of, float* __restrict__ out) {
  long t = blockIdx.x;
  long r0 = rows_of[t * 2], r1 = rows_of[t * 2 + 1];
  for (int i = threadIdx.x; i < HD; i += 256)
    out[t * HD + i] = x1[t * HD + i] + eo[r0 * HD + i] + eo[r1 * HD + i];
}

// ---------------------------------------------------------------------------
// Host: carve workspace (~510 MB), launch pipeline (graph-capture safe)
// ---------------------------------------------------------------------------
extern "C" void kernel_launch(void* const* d_in, const int* in_sizes, int n_in,
                              void* d_out, int out_size, void* d_ws, size_t ws_size,
                              hipStream_t stream) {
  const float* x     = (const float*)d_in[0];
  const float* ln1w  = (const float*)d_in[1];
  const float* ln1b  = (const float*)d_in[2];
  const float* ln2w  = (const float*)d_in[3];
  const float* ln2b  = (const float*)d_in[4];
  const float* in_w  = (const float*)d_in[5];
  const float* in_b  = (const float*)d_in[6];
  const float* out_w = (const float*)d_in[7];
  const float* out_b = (const float*)d_in[8];
  const float* gw    = (const float*)d_in[9];
  const float* gb    = (const float*)d_in[10];
  const float* w1    = (const float*)d_in[11];
  const float* b1    = (const float*)d_in[12];
  const float* w2    = (const float*)d_in[13];
  const float* b2    = (const float*)d_in[14];
  float* out = (float*)d_out;

  char* p = (char*)d_ws;
  auto carve = [&](size_t bytes) -> void* {
    void* r = (void*)p;
    p += (bytes + 255) & ~(size_t)255;
    return r;
  };
  bf16*  xn      = (bf16*)carve((size_t)TOK * HD * 2);
  bf16*  qkv     = (bf16*)carve((size_t)TOK * TH3 * 2);
  bf16*  ctx     = (bf16*)carve((size_t)TOK * HD * 2);
  float* x1      = (float*)carve((size_t)TOK * HD * 4);
  bf16*  xn2     = (bf16*)carve((size_t)TOK * HD * 2);
  bf16*  wq      = (bf16*)carve((size_t)TH3 * HD * 2);
  bf16*  wo      = (bf16*)carve((size_t)HD * HD * 2);
  bf16*  w1t     = (bf16*)carve((size_t)NE * HD * FD * 2);
  bf16*  w2t     = (bf16*)carve((size_t)NE * HD * FD * 2);
  bf16*  Xg      = (bf16*)carve((size_t)CAP * HD * 2);
  bf16*  hid     = (bf16*)carve((size_t)CAP * FD * 2);
  float* eo      = (float*)carve((size_t)CAP * HD * 4);
  int*   tk_idx  = (int*)carve((size_t)TOK * 2 * 4);
  float* tk_p    = (float*)carve((size_t)TOK * 2 * 4);
  int*   rows_of = (int*)carve((size_t)TOK * 2 * 4);
  int*   row_tok = (int*)carve((size_t)CAP * 4);
  float* row_p   = (float*)carve((size_t)CAP * 4);
  int*   count   = (int*)carve(64);
  int*   off     = (int*)carve(64);
  int*   cursor  = (int*)carve(64);

  hipMemsetAsync(count, 0, NE * sizeof(int), stream);
  hipMemsetAsync(row_tok, 0, (size_t)CAP * sizeof(int), stream);

  // weight conversions (fp32 -> bf16, W stored as [N,K] row-major)
  f2b_kernel<<<3072, 256, 0, stream>>>(in_w, wq, (long)TH3 * HD);
  f2b_kernel<<<1024, 256, 0, stream>>>(out_w, wo, (long)HD * HD);
  f2b_tr_kernel<<<8192, 256, 0, stream>>>(w1, w1t, NE, HD, FD);  // -> [E][F][H]
  f2b_tr_kernel<<<8192, 256, 0, stream>>>(w2, w2t, NE, FD, HD);  // -> [E][H][F]

  // attention sub-block
  ln_kernel<<<TOK, 256, 0, stream>>>(x, ln1w, ln1b, xn);
  wmma_gemm<0, false><<<dim3(TH3 / 128, TOK / 128), 256, 0, stream>>>(
      xn, wq, in_b, nullptr, nullptr, nullptr, qkv, TH3, HD, nullptr, 0, 0);
  attn_kernel<<<dim3(Sseq / 128, NHh, 8), 256, 0, stream>>>(qkv, ctx);
  wmma_gemm<1, false><<<dim3(HD / 128, TOK / 128), 256, 0, stream>>>(
      ctx, wo, out_b, x, nullptr, x1, nullptr, HD, HD, nullptr, 0, 0);

  // MoE sub-block (top-2 only; identical to dense-all + top-k combine)
  ln_kernel<<<TOK, 256, 0, stream>>>(x1, ln2w, ln2b, xn2);
  gate_kernel<<<TOK, 256, 0, stream>>>(xn2, gw, gb, tk_idx, tk_p, count);
  scan_kernel<<<1, 1, 0, stream>>>(count, off, cursor);
  place_kernel<<<TOK / 256, 256, 0, stream>>>(tk_idx, tk_p, cursor, rows_of,
                                              row_tok, row_p);
  gather_kernel<<<CAP, 128, 0, stream>>>(xn2, row_tok, Xg);
  wmma_gemm<2, true><<<dim3(FD / 128, 64, NE), 256, 0, stream>>>(
      Xg, w1t, b1, nullptr, nullptr, nullptr, hid, FD, HD, off,
      (long)HD * FD, FD);
  wmma_gemm<3, true><<<dim3(HD / 128, 64, NE), 256, 0, stream>>>(
      hid, w2t, b2, nullptr, row_p, eo, nullptr, HD, FD, off,
      (long)HD * FD, HD);
  combine_kernel<<<TOK, 256, 0, stream>>>(x1, eo, rows_of, out);
}
